// BalSCL_BAL_16690242913043
// MI455X (gfx1250) — matrix operand order repaired and checked by reading, not compiled, not run on hardware
//
#include <hip/hip_runtime.h>
#include <hip/hip_bf16.h>

// ---------------- types ----------------
typedef __attribute__((ext_vector_type(16))) __bf16 v16bf;
typedef __attribute__((ext_vector_type(8)))  __bf16 v8bf;
typedef __attribute__((ext_vector_type(4)))  __bf16 v4bf;
typedef __attribute__((ext_vector_type(8)))  float  v8f;

// vector-of-4-int type matching the async-LDS builtin's parameter type
typedef int v4i_vec __attribute__((vector_size(16)));
typedef __attribute__((address_space(1))) v4i_vec as1_v4i;
typedef __attribute__((address_space(3))) v4i_vec as3_v4i;

// ---------------- problem constants ----------------
#define B2    8192           // 2*batch
#define DIM   1024           // feature dim
#define NCLS  1000           // classes
#define NCOL  (B2 + NCLS)    // 9192 valid columns
#define NPAD  9216           // padded columns (72 tiles of 128)
#define CSPLIT 8             // column splits (grid.y)
#define CT_PER (NPAD / 128 / CSPLIT)  // 9 col tiles per split
#define INV_T 10.0f          // 1 / temperature

// ---------------- async global->LDS (CDNA5) with sync fallback ----------------
#if defined(__has_builtin)
#if __has_builtin(__builtin_amdgcn_global_load_async_to_lds_b128) && \
    __has_builtin(__builtin_amdgcn_s_wait_asynccnt)
#define HAVE_ASYNC_LDS 1
#endif
#endif
#ifndef HAVE_ASYNC_LDS
#define HAVE_ASYNC_LDS 0
#endif

__device__ __forceinline__ void copy16_to_lds(const __bf16* g, __bf16* l) {
#if HAVE_ASYNC_LDS
  __builtin_amdgcn_global_load_async_to_lds_b128((as1_v4i*)g, (as3_v4i*)l, 0, 0);
#else
  *(uint4*)l = *(const uint4*)g;
#endif
}
__device__ __forceinline__ void wait_async_all() {
#if HAVE_ASYNC_LDS
  __builtin_amdgcn_s_wait_asynccnt(0);
#endif
}

// ---------------- small setup kernels ----------------
__global__ void k_init(int* __restrict__ counts, float* __restrict__ out) {
  int i = blockIdx.x * blockDim.x + threadIdx.x;
  if (i < NCLS) counts[i] = 0;
  if (i == 0) out[0] = 0.0f;
}

__global__ void k_hist(const int* __restrict__ targets, int* __restrict__ counts) {
  int i = blockIdx.x * blockDim.x + threadIdx.x;
  if (i < B2) atomicAdd(&counts[targets[i]], 1);
}

// Per-column label + reciprocal weights: counts[c] = hist[c] + 1 (center).
__global__ void k_cols(const int* __restrict__ targets, const int* __restrict__ counts,
                       float* __restrict__ colr0, float* __restrict__ colr1,
                       int* __restrict__ colt) {
  int j = blockIdx.x * blockDim.x + threadIdx.x;
  if (j >= NPAD) return;
  if (j < NCOL) {
    int t = (j < B2) ? targets[j] : (j - B2);
    float cw = (float)(counts[t] + 1);
    colt[j]  = t;
    colr0[j] = 1.0f / cw;
    colr1[j] = (cw > 1.5f) ? (1.0f / (cw - 1.0f)) : 1.0f;
  } else {
    colt[j] = -1; colr0[j] = 1.0f; colr1[j] = 1.0f;
  }
}

// fp32 -> bf16 pack of [features ; centers1 ; zero-pad] into [NPAD][DIM]
__global__ void k_convert(const float* __restrict__ features,
                          const float* __restrict__ centers,
                          __bf16* __restrict__ feats) {
  size_t idx = ((size_t)blockIdx.x * blockDim.x + threadIdx.x) * 4;
  if (idx >= (size_t)NPAD * DIM) return;
  int row = (int)(idx / DIM);
  int col = (int)(idx % DIM);
  float4 v;
  if (row < B2)        v = *(const float4*)(features + (size_t)row * DIM + col);
  else if (row < NCOL) v = *(const float4*)(centers + (size_t)(row - B2) * DIM + col);
  else                 v = make_float4(0.f, 0.f, 0.f, 0.f);
  v4bf o = { (__bf16)v.x, (__bf16)v.y, (__bf16)v.z, (__bf16)v.w };
  *(v4bf*)(feats + idx) = o;
}

// ---------------- fused GEMM + online softmax-stats kernel ----------------
// Block: 256 threads (8 waves). Tile: 128 rows x 128 cols, K chunks of 32,
// double-buffered async global->LDS staging. Wave grid 4x2: wave (wm,wn)
// owns a 32x64 region = 2x4 WMMA accumulators.
__launch_bounds__(256)
__global__ void k_gemm_stats(const __bf16* __restrict__ feats,
                             const int*    __restrict__ targets,
                             const float*  __restrict__ colr0,
                             const float*  __restrict__ colr1,
                             const int*    __restrict__ colt,
                             float* __restrict__ pm, float* __restrict__ ps,
                             float* __restrict__ pp, float* __restrict__ pn) {
  constexpr int LDA = 40;   // padded LDS row stride (halves): 80B, 16B-aligned rows
  constexpr int LDC = 33;   // odd f32 stride -> conflict-free banks
  __shared__ __align__(16) __bf16 Abuf[2 * 128 * LDA];   // 20,480 B
  __shared__ __align__(16) __bf16 Bbuf[2 * 128 * LDA];   // 20,480 B
  __shared__ float Ctile[128 * LDC];                     // 16,896 B
  __shared__ float m_run[128], S_run[128], P_run[128], n_run[128];
  __shared__ float hm[256], hs[256], hp[256], hn[256];   // total LDS = 64,000 B

  const int tid  = threadIdx.x;
  const int lane = tid & 31;
  const int wave = tid >> 5;
  const int wm   = wave & 3;    // 0..3 -> 32-row group
  const int wn   = wave >> 2;   // 0..1 -> 64-col group
  const int l16  = lane & 15;
  const int hi   = lane >> 4;   // lane half
  const int rowbase = blockIdx.x * 128;

  // async staging: each thread moves 32B per tile (2 x b128) for A and B
  const int sr = tid >> 1, sh = tid & 1;
  auto issue = [&](int nq, int nkc) {
    const size_t gofs = (size_t)(nkc * 32 + sh * 16);
    const __bf16* gA = feats + (size_t)(rowbase + sr) * DIM + gofs;
    const __bf16* gB = feats + (size_t)((blockIdx.y * CT_PER + nq) * 128 + sr) * DIM + gofs;
    __bf16* lA = Abuf + (nkc & 1) * (128 * LDA) + sr * LDA + sh * 16;
    __bf16* lB = Bbuf + (nkc & 1) * (128 * LDA) + sr * LDA + sh * 16;
    copy16_to_lds(gA,     lA);
    copy16_to_lds(gA + 8, lA + 8);
    copy16_to_lds(gB,     lB);
    copy16_to_lds(gB + 8, lB + 8);
  };

  if (tid < 128) { m_run[tid] = -3.0e38f; S_run[tid] = 0.f; P_run[tid] = 0.f; n_run[tid] = 0.f; }

  issue(0, 0);  // prologue: prefetch first chunk into buffer 0

  for (int q = 0; q < CT_PER; ++q) {
    const int colbase = (blockIdx.y * CT_PER + q) * 128;

    v8f acc[2][4];
    #pragma unroll
    for (int a = 0; a < 2; ++a)
      #pragma unroll
      for (int b = 0; b < 4; ++b)
        #pragma unroll
        for (int e = 0; e < 8; ++e) acc[a][b][e] = 0.0f;

    // ---- K loop: double-buffered, one barrier per chunk, 8 WMMA per chunk ----
    for (int kc = 0; kc < DIM / 32; ++kc) {
      wait_async_all();      // own prefetched chunk has landed in LDS
      __syncthreads();       // everyone's chunk visible; prev buffer reads done

      const __bf16* At = Abuf + (kc & 1) * (128 * LDA);
      const __bf16* Bt = Bbuf + (kc & 1) * (128 * LDA);

      v16bf afr[2], bfr[4];
      #pragma unroll
      for (int a = 0; a < 2; ++a) {                    // A frag: 16x32, lane row M=l16
        const __bf16* ap = At + (wm * 32 + a * 16 + l16) * LDA + hi * 8;
        union { v16bf v; v8bf h[2]; } u;
        u.h[0] = *(const v8bf*)ap;
        u.h[1] = *(const v8bf*)(ap + 16);
        afr[a] = u.v;
      }
      #pragma unroll
      for (int b = 0; b < 4; ++b) {                    // B frag: 32x16, lane col N=l16
        const __bf16* bp = Bt + (wn * 64 + b * 16 + l16) * LDA + hi * 8;
        union { v16bf v; v8bf h[2]; } u;
        u.h[0] = *(const v8bf*)bp;
        u.h[1] = *(const v8bf*)(bp + 16);
        bfr[b] = u.v;
      }

      // prefetch next chunk (crosses q-boundary; overlaps with stats phase too)
      {
        int nq = q, nkc = kc + 1;
        if (nkc == DIM / 32) { nkc = 0; nq = q + 1; }
        if (nq < CT_PER) issue(nq, nkc);
      }

      #pragma unroll
      for (int a = 0; a < 2; ++a)
        #pragma unroll
        for (int b = 0; b < 4; ++b)
          acc[a][b] = __builtin_amdgcn_wmma_f32_16x16x32_bf16(
              false, afr[a], false, bfr[b], (short)0, acc[a][b], false, false);
    }

    // ---- online stats update, one 32-column quarter at a time ----
    #pragma unroll 1
    for (int qq = 0; qq < 4; ++qq) {
      __syncthreads();
      if (wn == (qq >> 1)) {
        const int bbase = (qq & 1) * 2;
        #pragma unroll
        for (int a = 0; a < 2; ++a)
          #pragma unroll
          for (int bb = 0; bb < 2; ++bb) {
            int r0 = wm * 32 + a * 16 + hi * 8;
            int c  = bb * 16 + l16;
            #pragma unroll
            for (int e = 0; e < 8; ++e) Ctile[(r0 + e) * LDC + c] = acc[a][bbase + bb][e];
          }
      }
      __syncthreads();

      const int r = tid >> 1, ch = tid & 1;            // 2 threads per row, 16 cols each
      const int row_g = rowbase + r;
      const int tgt   = targets[row_g];
      const int colg0 = colbase + qq * 32 + ch * 16;
      const float* crow = Ctile + r * LDC + ch * 16;

      float lmax = -3.0e38f;
      for (int cc = 0; cc < 16; ++cc)
        if (colg0 + cc < NCOL) lmax = fmaxf(lmax, crow[cc] * INV_T);
      hm[tid] = lmax;
      __syncthreads();

      const float om    = m_run[r];
      const float m_new = fmaxf(om, fmaxf(hm[tid & ~1], hm[tid | 1]));
      float s = 0.f, p = 0.f, n = 0.f;
      for (int cc = 0; cc < 16; ++cc) {
        int cg = colg0 + cc;
        if (cg < NCOL && cg != row_g) {
          float l = crow[cc] * INV_T;
          float e = __expf(l - m_new);
          bool pos = (colt[cg] == tgt);
          s += e * (pos ? colr1[cg] : colr0[cg]);
          if (pos) { p += l; n += 1.f; }
        }
      }
      hs[tid] = s; hp[tid] = p; hn[tid] = n;
      __syncthreads();
      if (ch == 0) {
        float scale = (om > -1.0e38f) ? __expf(om - m_new) : 0.f;
        S_run[r] = S_run[r] * scale + hs[tid] + hs[tid + 1];
        P_run[r] += hp[tid] + hp[tid + 1];
        n_run[r] += hn[tid] + hn[tid + 1];
        m_run[r]  = m_new;
      }
    }
  }

  __syncthreads();
  if (tid < 128) {
    size_t o = (size_t)blockIdx.y * B2 + (rowbase + tid);
    pm[o] = m_run[tid]; ps[o] = S_run[tid]; pp[o] = P_run[tid]; pn[o] = n_run[tid];
  }
}

// ---------------- merge column-split partials -> scalar loss ----------------
__global__ void k_merge(const float* __restrict__ pm, const float* __restrict__ ps,
                        const float* __restrict__ pp, const float* __restrict__ pn,
                        float* __restrict__ out) {
  int i = blockIdx.x * blockDim.x + threadIdx.x;
  if (i >= B2) return;
  float m = -3.0e38f;
  for (int k = 0; k < CSPLIT; ++k) m = fmaxf(m, pm[(size_t)k * B2 + i]);
  float S = 0.f, P = 0.f, N = 0.f;
  for (int k = 0; k < CSPLIT; ++k) {
    float mk = pm[(size_t)k * B2 + i];
    if (mk > -1.0e38f) S += ps[(size_t)k * B2 + i] * __expf(mk - m);
    P += pp[(size_t)k * B2 + i];
    N += pn[(size_t)k * B2 + i];
  }
  // loss_i = -(P/N - m - log S) = m + log S - P/N
  float loss = m + __logf(S) - P / N;
  atomicAdd(out, loss * (1.0f / (float)B2));
}

// ---------------- launch ----------------
extern "C" void kernel_launch(void* const* d_in, const int* in_sizes, int n_in,
                              void* d_out, int out_size, void* d_ws, size_t ws_size,
                              hipStream_t stream) {
  const float* centers  = (const float*)d_in[0];   // [NCLS, DIM]
  const float* features = (const float*)d_in[1];   // [B2, DIM]
  const int*   targets  = (const int*)d_in[2];     // [B2]
  float* out = (float*)d_out;

  char* ws = (char*)d_ws;
  size_t o = 0;
  __bf16* feats = (__bf16*)(ws + o); o += (size_t)NPAD * DIM * 2;   // 18.9 MB
  int*   counts = (int*)(ws + o);    o += 4096;
  float* colr0  = (float*)(ws + o);  o += (size_t)NPAD * 4;
  float* colr1  = (float*)(ws + o);  o += (size_t)NPAD * 4;
  int*   colt   = (int*)(ws + o);    o += (size_t)NPAD * 4;
  float* pm = (float*)(ws + o); o += (size_t)CSPLIT * B2 * 4;
  float* ps = (float*)(ws + o); o += (size_t)CSPLIT * B2 * 4;
  float* pp = (float*)(ws + o); o += (size_t)CSPLIT * B2 * 4;
  float* pn = (float*)(ws + o); o += (size_t)CSPLIT * B2 * 4;

  k_init<<<4, 256, 0, stream>>>(counts, out);
  k_hist<<<B2 / 256, 256, 0, stream>>>(targets, counts);
  k_cols<<<NPAD / 256, 256, 0, stream>>>(targets, counts, colr0, colr1, colt);
  k_convert<<<((size_t)NPAD * DIM / 4 + 255) / 256, 256, 0, stream>>>(features, centers, feats);

  dim3 grid(B2 / 128, CSPLIT);
  k_gemm_stats<<<grid, 256, 0, stream>>>(feats, targets, colr0, colr1, colt, pm, ps, pp, pn);
  k_merge<<<B2 / 256, 256, 0, stream>>>(pm, ps, pp, pn, out);
}
